// CharacterClassificationModel_52192442581105
// MI455X (gfx1250) — matrix operand order repaired and checked by reading, not compiled, not run on hardware
//
#include <hip/hip_runtime.h>

typedef _Float16 f16;
typedef __attribute__((ext_vector_type(16))) _Float16 v16h;
typedef __attribute__((ext_vector_type(8)))  _Float16 v8h;
typedef __attribute__((ext_vector_type(8)))  float    v8f;
typedef unsigned int u32x4 __attribute__((ext_vector_type(4)));
typedef int          i32x4 __attribute__((ext_vector_type(4)));
typedef int          i32x8 __attribute__((ext_vector_type(8)));

#define TT    4096
#define HH    512
#define EE    16
#define OO    64
#define GATES 2048        // 4*H
#define KCH   16          // H/32 k-chunks
#define TBLK  256         // T/16 time blocks

// ---------------- workspace layout (bytes) ----------------
static constexpr size_t OFF_XA    = 0;                                   // 256 A-tiles * 512 halves
static constexpr size_t SZ_XA     = (size_t)TBLK * 512 * 2;
static constexpr size_t SZ_WDIR   = (size_t)(128 + 3*2048 + 4*2048) * 512 * 2; // 14,811,136
static constexpr size_t OFF_WD0   = OFF_XA + SZ_XA;
static constexpr size_t OFF_WD1   = OFF_WD0 + SZ_WDIR;
static constexpr size_t OFF_W1    = OFF_WD1 + SZ_WDIR;
static constexpr size_t SZ_W1     = (size_t)1024 * 512 * 2;
static constexpr size_t OFF_HST   = OFF_W1 + SZ_W1;
static constexpr size_t SZ_HST    = (size_t)2 * 4 * TT * HH * 2;          // f16 h storage
static constexpr size_t OFF_CST   = OFF_HST + SZ_HST;
static constexpr size_t OFF_BSUM  = OFF_CST + SZ_HST;
static constexpr size_t SZ_BSUM   = (size_t)2 * 4 * GATES * 4;
static constexpr size_t OFF_FLAGS = OFF_BSUM + SZ_BSUM;

__host__ __device__ constexpr size_t wih_tile_off(int l) {
  return (l == 0) ? 0 : (size_t)(128 + (l - 1) * 2048);
}
__host__ __device__ constexpr size_t whh_tile_off(int l) {
  return (size_t)(128 + 3 * 2048) + (size_t)l * 2048;
}

__device__ __forceinline__ v8f wmma_f16(v16h a, v16h b, v8f c) {
  // D(16x16,f32) = A(16x32,f16) * B(32x16,f16) + C
  return __builtin_amdgcn_wmma_f32_16x16x32_f16(false, a, false, b, (short)0, c,
                                                false, false);
}

// ---- Tensor Data Mover: 16KB contiguous global -> LDS (1-row f16 tensor) ----
// D# built per CDNA5 ISA:
//  group0: count=1 | lds_addr | global_addr[56:0] | type=2
//  group1: data_size=2B, tensor_dim0=tile_dim0=8192, tensor_dim1=tile_dim1=1,
//          tensor_dim0_stride=8192, no mask/pad/iterate/barrier
// 6-arg builtin form (clang-23 / therock-10.0 headers): (g0, g1, g2, g3, g4, cpol)
__device__ __forceinline__ void tdm_load_16kb(const void* gsrc, void* lds_dst) {
  unsigned long long ga = (unsigned long long)(size_t)gsrc;
  unsigned ldsa = (unsigned)(size_t)lds_dst;       // generic LDS addr[31:0] = LDS offset
  u32x4 g0;
  g0[0] = 1u;                                      // count=1 (valid user descriptor)
  g0[1] = ldsa;                                    // lds_addr
  g0[2] = (unsigned)(ga & 0xFFFFFFFFu);            // global_addr lo
  g0[3] = (unsigned)(ga >> 32) | (2u << 30);       // global_addr hi | type=2 (image)
  i32x8 g1;
  g1[0] = 0x00010000;                              // data_size=1 (2 bytes)
  g1[1] = 0x20000000;                              // tensor_dim0 = 8192 (bits 79:48 lo)
  g1[2] = 0x00010000;                              // tensor_dim1 = 1
  g1[3] = 0x20000000;                              // tile_dim0 = 8192
  g1[4] = 0x00000001;                              // tile_dim1 = 1, tile_dim2 = 0
  g1[5] = 8192;                                    // tensor_dim0_stride lo
  g1[6] = 0;
  g1[7] = 0;
  i32x4 z4 = {0, 0, 0, 0};
  i32x8 z8 = {0, 0, 0, 0, 0, 0, 0, 0};
  __builtin_amdgcn_tensor_load_to_lds(g0, g1, z4, z4, z8, 0);
}

// ---------------- weight repack: row-major fp32 W[n][k] -> WMMA B-tiles f16 ----
// B layout (32x16 f16): lane holds column n = lane&15; khalf=(lane>>4)*8; e<8 -> k=khalf+e ; e>=8 -> k=16+khalf+(e-8).
__global__ void pack_b_kernel(const float* __restrict__ W, f16* __restrict__ dst,
                              int K, int KC, long totalHalves) {
  long idx = (long)blockIdx.x * blockDim.x + threadIdx.x;
  if (idx >= totalHalves) return;
  int  e    = (int)(idx & 15);
  int  lane = (int)((idx >> 4) & 31);
  long tile = idx >> 9;
  long nT   = tile / KC;
  long kc   = tile % KC;
  int  n    = (int)(nT * 16 + (lane & 15));
  int  k    = (int)(kc * 32 + ((e < 8) ? ((lane >> 4) * 8 + e) : (16 + (lane >> 4) * 8 + (e - 8))));
  float v   = (k < K) ? W[(size_t)n * K + k] : 0.f;
  dst[idx] = (f16)v;
}

// ---------------- embedding -> WMMA A-tiles (16 timesteps x K=32, E padded) ----
// A layout (16x32 f16): lane holds row m = lane&15; khalf=(lane>>4)*8;
// e<8 -> k=khalf+e ; e>=8 -> k=16+khalf+(e-8).
__global__ void pack_x_kernel(const int* __restrict__ tokens,
                              const float* __restrict__ embed,
                              f16* __restrict__ dst) {
  long idx = (long)blockIdx.x * blockDim.x + threadIdx.x;
  if (idx >= (long)TBLK * 512) return;
  int  e    = (int)(idx & 15);
  int  lane = (int)((idx >> 4) & 31);
  long tile = idx >> 9;
  int  m     = lane & 15;
  int  t     = (int)tile * 16 + m;
  int  khalf = (lane >> 4) * 8;
  int  k     = (e < 8) ? (khalf + e) : (16 + khalf + (e - 8));
  float v = (k < EE) ? embed[(size_t)tokens[t] * EE + k] : 0.f;
  dst[idx] = (f16)v;
}

__global__ void bias_sum_kernel(const float* __restrict__ fbi, const float* __restrict__ fbh,
                                const float* __restrict__ bbi, const float* __restrict__ bbh,
                                float* __restrict__ outb) {
  int idx = blockIdx.x * blockDim.x + threadIdx.x;
  if (idx >= 2 * 4 * GATES) return;
  int d = idx >> 13, r = idx & 8191;
  outb[idx] = d ? (bbi[r] + bbh[r]) : (fbi[r] + fbh[r]);
}

__global__ void init_flags_kernel(int* flags) {
  if (threadIdx.x < 8) flags[threadIdx.x] = 0;
}

// ---------------- persistent bidirectional pipelined LSTM ----------------
// grid = 8 blocks: block -> (dir = bx>>2, layer = bx&3). 512 threads = 16 waves.
// Each wave owns 8 gate n-tiles (16 waves * 8 * 16 = 2048 gates).
__global__ __launch_bounds__(512) void lstm_kernel(char* __restrict__ ws) {
  const int bx = blockIdx.x;
  const int d = bx >> 2, l = bx & 3;

  const f16*  xA   = (const f16*)(ws + OFF_XA);
  const f16*  wdir = (const f16*)(ws + (d ? OFF_WD1 : OFF_WD0));
  const f16*  wih  = wdir + wih_tile_off(l) * 512;
  const f16*  whh  = wdir + whh_tile_off(l) * 512;
  const float* bsum = (const float*)(ws + OFF_BSUM) + (size_t)(d * 4 + l) * GATES;
  f16* hst = (f16*)(ws + OFF_HST);
  f16* cst = (f16*)(ws + OFF_CST);
  f16* hOut = hst + (size_t)(d * 4 + l) * TT * HH;
  f16* cOut = cst + (size_t)(d * 4 + l) * TT * HH;
  const f16* hPrev = (l > 0) ? (hst + (size_t)(d * 4 + l - 1) * TT * HH) : (const f16*)0;
  int* flags   = (int*)(ws + OFF_FLAGS);
  int* myFlag  = flags + (d * 4 + l);
  int* srcFlag = flags + (d * 4 + l - 1);

  const int tid = threadIdx.x, lane = tid & 31, wv = tid >> 5;
  const int nlo = lane & 15, khalf = (lane >> 4) * 8;

  __shared__ f16 sm_gin[16 * GATES];                 // 64KB: per-block input contributions
  __shared__ __align__(128) f16 sm_stage[16 * HH];   // 16KB: phase1 hprev stage / phase2 gates (alias)
  __shared__ __align__(16) f16 sm_h[HH];
  __shared__ float sm_c[HH];
  float* gatesF = (float*)sm_stage;

  if (tid < HH) { sm_h[tid] = (f16)0.f; sm_c[tid] = 0.f; }
  __syncthreads();

  for (int p = 0; p < TBLK; ++p) {
    const int tb = d ? (TBLK - 1 - p) : p;         // backward walks blocks in reverse
    const size_t tbase = (size_t)tb * 16;

    if (l > 0) {
      if (tid == 0)
        while (__hip_atomic_load(srcFlag, __ATOMIC_ACQUIRE, __HIP_MEMORY_SCOPE_AGENT) <= p)
          __builtin_amdgcn_s_sleep(2);
      __syncthreads();
      // stage previous layer's 16x512 h block via Tensor Data Mover (16KB contiguous)
      if (wv == 0) {
        tdm_load_16kb(hPrev + tbase * HH, sm_stage);
        __builtin_amdgcn_s_wait_tensorcnt(0);
      }
      __syncthreads();
    }

    // ---- Phase 1: Gin[16 x 2048] = inp_block @ W_ih^T + (b_ih+b_hh) -- dense WMMA GEMM
    v8f accG[8];
    #pragma unroll
    for (int t8 = 0; t8 < 8; ++t8) {
      float bv = bsum[(wv * 8 + t8) * 16 + nlo];
      v8f a;
      #pragma unroll
      for (int r = 0; r < 8; ++r) a[r] = bv;       // C[m][n] = bias[n]
      accG[t8] = a;
    }
    if (l == 0) {
      v16h A = *(const v16h*)(xA + (size_t)tb * 512 + lane * 16);  // K=32 padded embed
      #pragma unroll
      for (int t8 = 0; t8 < 8; ++t8) {
        const v16h* bp = (const v16h*)(wih + (size_t)(wv * 8 + t8) * 512);
        accG[t8] = wmma_f16(A, bp[lane], accG[t8]);
      }
    } else {
      for (int c = 0; c < KCH; ++c) {
        v8h lo = *(const v8h*)(sm_stage + nlo * HH + c * 32 + khalf);
        v8h hi = *(const v8h*)(sm_stage + nlo * HH + c * 32 + 16 + khalf);
        v16h A;
        #pragma unroll
        for (int i = 0; i < 8; ++i) { A[i] = lo[i]; A[i + 8] = hi[i]; }
        #pragma unroll
        for (int t8 = 0; t8 < 8; ++t8) {
          const v16h* bp = (const v16h*)(wih + (size_t)((wv * 8 + t8) * KCH + c) * 512);
          accG[t8] = wmma_f16(A, bp[lane], accG[t8]);
        }
      }
    }
    #pragma unroll
    for (int t8 = 0; t8 < 8; ++t8)
      #pragma unroll
      for (int r = 0; r < 8; ++r)
        sm_gin[(size_t)(r + (lane >> 4) * 8) * GATES + (wv * 8 + t8) * 16 + nlo] =
            (f16)accG[t8][r];
    __syncthreads();

    // ---- Phase 2: 16 sequential recurrent steps (h @ W_hh^T via row-replicated A)
    for (int s = 0; s < 16; ++s) {
      const int m = d ? (15 - s) : s;              // ascending-t index within block
      const size_t t = tbase + m;
      v8f acc[8];
      #pragma unroll
      for (int t8 = 0; t8 < 8; ++t8) {
        float g0 = (float)sm_gin[(size_t)m * GATES + (wv * 8 + t8) * 16 + nlo];
        v8f a;
        #pragma unroll
        for (int r = 0; r < 8; ++r) a[r] = g0;
        acc[t8] = a;
      }
      for (int c = 0; c < KCH; ++c) {
        // every A row = h chunk  =>  every lane ends with gates[n=lane&15] in acc[0]
        v8h lo = *(const v8h*)(sm_h + c * 32 + khalf);
        v8h hi = *(const v8h*)(sm_h + c * 32 + 16 + khalf);
        v16h A;
        #pragma unroll
        for (int i = 0; i < 8; ++i) { A[i] = lo[i]; A[i + 8] = hi[i]; }
        #pragma unroll
        for (int t8 = 0; t8 < 8; ++t8) {
          const v16h* bp = (const v16h*)(whh + (size_t)((wv * 8 + t8) * KCH + c) * 512);
          acc[t8] = wmma_f16(A, bp[lane], acc[t8]);
        }
      }
      if (lane < 16) {
        #pragma unroll
        for (int t8 = 0; t8 < 8; ++t8)
          gatesF[(wv * 8 + t8) * 16 + lane] = acc[t8][0];
      }
      __syncthreads();
      if (tid < HH) {
        const int j = tid;
        float gi = gatesF[j], gf = gatesF[HH + j];
        float gg = gatesF[2 * HH + j], go = gatesF[3 * HH + j];
        float si = 1.f / (1.f + __expf(-gi));
        float sf = 1.f / (1.f + __expf(-gf));
        float so = 1.f / (1.f + __expf(-go));
        float cn = sf * sm_c[j] + si * tanhf(gg);
        float hn = so * tanhf(cn);
        sm_c[j] = cn;
        sm_h[j] = (f16)hn;
        hOut[t * HH + j] = (f16)hn;
        cOut[t * HH + j] = (f16)cn;
      }
      __syncthreads();
    }

    __threadfence();                                // flush this thread's global stores
    __syncthreads();
    if (tid == 0)
      __hip_atomic_store(myFlag, p + 1, __ATOMIC_RELEASE, __HIP_MEMORY_SCOPE_AGENT);
  }
}

// ---------------- FC (feat @ W1^T + b1) + softmax, WMMA GEMM ----------------
// feat[t] = [hf0,hb0,hf1,hb1,hf2,hb2,hf3,hb3, cf0,cb0,...,cf3,cb3], 16*512 = 8192
// Feature segments are double-buffered through LDS with async TDM loads:
// wave 0 issues segment s+1 while all waves run WMMA on segment s.
__global__ __launch_bounds__(128) void fc_softmax_kernel(const char* __restrict__ ws,
                                                         const float* __restrict__ b1,
                                                         float* __restrict__ out) {
  const int blk = blockIdx.x;                       // 256 time tiles
  const int tid = threadIdx.x, lane = tid & 31, wv = tid >> 5;  // 4 waves, 1 o-tile each
  const int nlo = lane & 15, khalf = (lane >> 4) * 8;
  const f16* hst = (const f16*)(ws + OFF_HST);
  const f16* cst = (const f16*)(ws + OFF_CST);
  const f16* w1  = (const f16*)(ws + OFF_W1);
  __shared__ __align__(128) f16 sm_feat[2][16 * HH];  // 2 x 16KB double buffer
  __shared__ float sm_logits[16 * OO];

  const size_t tbase = (size_t)blk * 16;
  auto seg_src = [&](int seg) -> const f16* {
    const int j = seg >> 1, dir = seg & 1;
    return (((j < 4) ? hst : cst) + (size_t)(dir * 4 + (j & 3)) * TT * HH) + tbase * HH;
  };

  v8f acc;
  {
    float bv = b1[wv * 16 + nlo];
    #pragma unroll
    for (int r = 0; r < 8; ++r) acc[r] = bv;
  }

  if (wv == 0) tdm_load_16kb(seg_src(0), sm_feat[0]);   // prologue

  for (int seg = 0; seg < 16; ++seg) {
    if (wv == 0) {
      if (seg < 15) {
        tdm_load_16kb(seg_src(seg + 1), sm_feat[(seg + 1) & 1]);
        __builtin_amdgcn_s_wait_tensorcnt(1);  // in-order: seg's load done, seg+1 in flight
      } else {
        __builtin_amdgcn_s_wait_tensorcnt(0);
      }
    }
    __syncthreads();
    const f16* fseg = sm_feat[seg & 1];
    for (int c = 0; c < KCH; ++c) {
      v8h lo = *(const v8h*)(fseg + nlo * HH + c * 32 + khalf);
      v8h hi = *(const v8h*)(fseg + nlo * HH + c * 32 + 16 + khalf);
      v16h A;
      #pragma unroll
      for (int i = 0; i < 8; ++i) { A[i] = lo[i]; A[i + 8] = hi[i]; }
      const v16h* bp = (const v16h*)(w1 + (size_t)(wv * 256 + seg * 16 + c) * 512);
      acc = wmma_f16(A, bp[lane], acc);
    }
    __syncthreads();   // compute done before this buffer is overwritten (seg+2 issue)
  }
  #pragma unroll
  for (int r = 0; r < 8; ++r)
    sm_logits[(r + (lane >> 4) * 8) * OO + wv * 16 + nlo] = acc[r];
  __syncthreads();
  if (tid < 16) {
    const int row = tid;
    float mx = -3.4e38f;
    for (int o = 0; o < OO; ++o) mx = fmaxf(mx, sm_logits[row * OO + o]);
    float s = 0.f;
    for (int o = 0; o < OO; ++o) s += __expf(sm_logits[row * OO + o] - mx);
    float inv = 1.f / s;
    for (int o = 0; o < OO; ++o)
      out[(tbase + row) * OO + o] = __expf(sm_logits[row * OO + o] - mx) * inv;
  }
}

// ---------------- host side ----------------
extern "C" void kernel_launch(void* const* d_in, const int* in_sizes, int n_in,
                              void* d_out, int out_size, void* d_ws, size_t ws_size,
                              hipStream_t stream) {
  (void)in_sizes; (void)n_in; (void)out_size; (void)ws_size;
  const int*   tokens = (const int*)d_in[0];
  const float* embed  = (const float*)d_in[1];
  const float* fW_ih0 = (const float*)d_in[2];
  const float* fW_ih  = (const float*)d_in[3];
  const float* fW_hh  = (const float*)d_in[4];
  const float* fb_ih  = (const float*)d_in[5];
  const float* fb_hh  = (const float*)d_in[6];
  const float* bW_ih0 = (const float*)d_in[7];
  const float* bW_ih  = (const float*)d_in[8];
  const float* bW_hh  = (const float*)d_in[9];
  const float* bb_ih  = (const float*)d_in[10];
  const float* bb_hh  = (const float*)d_in[11];
  const float* W1     = (const float*)d_in[12];
  const float* b1     = (const float*)d_in[13];
  char* ws = (char*)d_ws;
  float* out = (float*)d_out;

  // embedding -> A tiles
  pack_x_kernel<<<(TBLK * 512 + 255) / 256, 256, 0, stream>>>(
      tokens, embed, (f16*)(ws + OFF_XA));

  const float* Wih0[2] = {fW_ih0, bW_ih0};
  const float* Wih[2]  = {fW_ih, bW_ih};
  const float* Whh[2]  = {fW_hh, bW_hh};
  for (int d = 0; d < 2; ++d) {
    f16* base = (f16*)(ws + (d ? OFF_WD1 : OFF_WD0));
    {
      long halves = (long)128 * 512;                 // W_ih0: N=2048, K=16 (pad 32), KC=1
      pack_b_kernel<<<(halves + 255) / 256, 256, 0, stream>>>(
          Wih0[d], base + wih_tile_off(0) * 512, EE, 1, halves);
    }
    for (int l = 1; l < 4; ++l) {
      long halves = (long)2048 * 512;                // N=2048, K=512, KC=16
      pack_b_kernel<<<(halves + 255) / 256, 256, 0, stream>>>(
          Wih[d] + (size_t)(l - 1) * GATES * HH, base + wih_tile_off(l) * 512,
          HH, KCH, halves);
    }
    for (int l = 0; l < 4; ++l) {
      long halves = (long)2048 * 512;
      pack_b_kernel<<<(halves + 255) / 256, 256, 0, stream>>>(
          Whh[d] + (size_t)l * GATES * HH, base + whh_tile_off(l) * 512,
          HH, KCH, halves);
    }
  }
  {
    long halves = (long)1024 * 512;                  // W1: N=64, K=8192, KC=256
    pack_b_kernel<<<(halves + 255) / 256, 256, 0, stream>>>(
        W1, (f16*)(ws + OFF_W1), 8192, 256, halves);
  }
  bias_sum_kernel<<<(2 * 4 * GATES + 255) / 256, 256, 0, stream>>>(
      fb_ih, fb_hh, bb_ih, bb_hh, (float*)(ws + OFF_BSUM));
  init_flags_kernel<<<1, 32, 0, stream>>>((int*)(ws + OFF_FLAGS));

  // pipelined bidirectional LSTM: 8 persistent WGs = (2 dirs x 4 layers)
  lstm_kernel<<<8, 512, 0, stream>>>(ws);

  // FC + softmax
  fc_softmax_kernel<<<TBLK, 128, 0, stream>>>(ws, b1, out);
}
